// CodecDecoder_81114752352311
// MI455X (gfx1250) — compile-verified, matrix operand order checked by reading, not run on hardware
//
#include <hip/hip_runtime.h>
#include <hip/hip_bf16.h>

#define DIMC 1024
#define NH 8
#define HDIM 128
#define HIDC 4096
#define EPSV 0.01f
#define SEMN 8192
#define SEMD 256
#define NACC 36
#define PATCHC 240
#define BATCH 2
#define TT0 256

typedef __bf16 v16bf  __attribute__((ext_vector_type(16)));
typedef __bf16 bf16x8 __attribute__((ext_vector_type(8)));
typedef float  v8f    __attribute__((ext_vector_type(8)));

__device__ __forceinline__ unsigned short f2bf_bits(float f) {
  unsigned u = __builtin_bit_cast(unsigned, f);
  u += 0x7FFFu + ((u >> 16) & 1u);           // round-to-nearest-even
  return (unsigned short)(u >> 16);
}
__device__ __forceinline__ __bf16 f2bf(float f) {
  unsigned short s = f2bf_bits(f);
  return __builtin_bit_cast(__bf16, s);
}
__device__ __forceinline__ float bf2f(__bf16 x) {
  unsigned short s = __builtin_bit_cast(unsigned short, x);
  unsigned u = ((unsigned)s) << 16;
  return __builtin_bit_cast(float, u);
}

// ---------------------------------------------------------------------------
// bf16 WMMA GEMM with async Global->LDS double-buffered staging.
//   C(M,N) = A(M,K) * W(N,K)^T ; A,W bf16 row-major; C fp32 or bf16.
// block = 256 threads = 8 waves; wave tile 16x64; block tile 32x256; K step 32.
// Per K-step the block stages:
//   W tile: 256 cols x 32 K  = 16 KB   (each col = 64 contiguous bytes)
//   A tile:  32 rows x 32 K  =  2 KB
// via GLOBAL_LOAD_ASYNC_TO_LDS_B128/B64 (5 async ops per thread per step),
// double buffered; s_wait_asynccnt + workgroup barrier fence the pipeline.
// Fragments (ISA 16-bit 16x32 layout: lane -> m=lane%16, kh=lane/16, two
// 16-byte K-chunks) are then read with ds_load_b128.
// ---------------------------------------------------------------------------
#define WTILE_BYTES (256 * 64)
#define ATILE_BYTES (32 * 64)
#define BUF_BYTES   (WTILE_BYTES + ATILE_BYTES)

template <int OUT_BF>
__global__ void __launch_bounds__(256)
k_gemm_wmma(const __bf16* __restrict__ A, const __bf16* __restrict__ W,
            void* __restrict__ Cout, int M, int N, int K) {
  __shared__ __align__(16) char lds[2 * BUF_BYTES];

  const int tid  = threadIdx.x;
  const int lane = tid & 31;
  const int wave = tid >> 5;
  const int wm   = wave >> 2;      // 0..1
  const int wn   = wave & 3;       // 0..3
  const int mBase = blockIdx.y * 32 + wm * 16;
  const int nBase = blockIdx.x * 256 + wn * 64;
  const int l16 = lane & 15;
  const int kh  = lane >> 4;

  // staging assignments (uniform across the block; EXEC all-ones throughout)
  const int wcol = tid >> 2;       // base col 0..63, +64 per rep
  const int wchk = tid & 3;        // 16B chunk within a 64B col-row
  const int arow = tid >> 3;       // 0..31
  const int achk = tid & 7;        // 8B chunk within a 64B row
  const size_t gA = (size_t)(blockIdx.y * 32 + arow) * K + achk * 4;
  const size_t gWcolBase = (size_t)(blockIdx.x * 256);

  auto issue = [&](int k0, int buf) {
    const unsigned base = (unsigned)(uintptr_t)&lds[buf * BUF_BYTES];
#pragma unroll
    for (int rep = 0; rep < 4; ++rep) {
      const int col = rep * 64 + wcol;
      const __bf16* g = W + (gWcolBase + col) * K + k0 + wchk * 8;
      const unsigned l = base + col * 64 + wchk * 16;
      asm volatile("global_load_async_to_lds_b128 %0, %1, off"
                   :: "v"(l), "v"(g) : "memory");
    }
    {
      const __bf16* g = A + gA + k0;
      const unsigned l = base + WTILE_BYTES + arow * 64 + achk * 8;
      asm volatile("global_load_async_to_lds_b64 %0, %1, off"
                   :: "v"(l), "v"(g) : "memory");
    }
  };

  v8f acc[4];
#pragma unroll
  for (int i = 0; i < 4; ++i)
#pragma unroll
    for (int r = 0; r < 8; ++r) acc[i][r] = 0.0f;

  union V16 { v16bf v; bf16x8 h[2]; };

  const int nk = K >> 5;
  issue(0, 0);
  for (int i = 0; i < nk; ++i) {
    const int buf = i & 1;
    if (i + 1 < nk) {
      issue((i + 1) * 32, buf ^ 1);
      // batch for buf i was issued before batch buf i+1; async loads complete
      // in order, so cnt<=5 means buf i is fully landed.
      asm volatile("s_wait_asynccnt 0x5" ::: "memory");
    } else {
      asm volatile("s_wait_asynccnt 0x0" ::: "memory");
    }
    __syncthreads();                       // all waves' tile data visible

    const char* wb = &lds[buf * BUF_BYTES];
    const char* ab = wb + WTILE_BYTES;
    V16 a;
    a.h[0] = *(const bf16x8*)(ab + (wm * 16 + l16) * 64 + kh * 16);
    a.h[1] = *(const bf16x8*)(ab + (wm * 16 + l16) * 64 + 32 + kh * 16);
#pragma unroll
    for (int nt = 0; nt < 4; ++nt) {
      const int col = wn * 64 + nt * 16 + l16;
      V16 b;
      b.h[0] = *(const bf16x8*)(wb + col * 64 + kh * 16);
      b.h[1] = *(const bf16x8*)(wb + col * 64 + 32 + kh * 16);
      acc[nt] = __builtin_amdgcn_wmma_f32_16x16x32_bf16(
          false, a.v, false, b.v, (short)0, acc[nt], false, false);
    }
    __syncthreads();                       // reads done before buf reuse
  }

  // D layout: lane -> n = nTile + lane%16; VGPR r -> m = mBase + r + (lane/16)*8
#pragma unroll
  for (int nt = 0; nt < 4; ++nt) {
    const int col = nBase + nt * 16 + l16;
#pragma unroll
    for (int r = 0; r < 8; ++r) {
      const int row = mBase + kh * 8 + r;
      const float vv = acc[nt][r];
      if (OUT_BF)
        ((unsigned short*)Cout)[(size_t)row * N + col] = f2bf_bits(vv);
      else
        ((float*)Cout)[(size_t)row * N + col] = vv;
    }
  }
}

// ---------------------------------------------------------------------------
// RMSNorm over C=1024 (256 threads x 4), fp32 in, fp32 or bf16 out (in-place OK)
// ---------------------------------------------------------------------------
template <int OUT_BF>
__global__ void k_rms(const float* __restrict__ x, const float* __restrict__ w,
                      void* __restrict__ y, int C) {
  const int row = blockIdx.x;
  const float* xr = x + (size_t)row * C;
  float lv[4];
  float ss = 0.0f;
#pragma unroll
  for (int i = 0; i < 4; ++i) {
    lv[i] = xr[threadIdx.x + 256 * i];
    ss += lv[i] * lv[i];
  }
  __shared__ float red[256];
  red[threadIdx.x] = ss;
  __syncthreads();
  for (int off = 128; off; off >>= 1) {
    if ((int)threadIdx.x < off) red[threadIdx.x] += red[threadIdx.x + off];
    __syncthreads();
  }
  const float scale = rsqrtf(red[0] / (float)C + EPSV);
#pragma unroll
  for (int i = 0; i < 4; ++i) {
    const int c = threadIdx.x + 256 * i;
    const float o = lv[i] * scale * w[c];
    if (OUT_BF) ((unsigned short*)y)[(size_t)row * C + c] = f2bf_bits(o);
    else        ((float*)y)[(size_t)row * C + c] = o;
  }
}

// ---------------------------------------------------------------------------
// weight-norm for causal convs: v (O,Cin,kk), norm per o over Cin*kk.
// Output TRANSPOSED to (O, kk, Cin) so the conv inner loop is contiguous.
// ---------------------------------------------------------------------------
__global__ void k_wn_oc(const float* __restrict__ v, const float* __restrict__ g,
                        float* __restrict__ w, int Cin, int kk) {
  const int o = blockIdx.x;
  const int R = Cin * kk;
  const float* vr = v + (size_t)o * R;
  float ss = 0.0f;
  for (int i = threadIdx.x; i < R; i += 256) ss += vr[i] * vr[i];
  __shared__ float red[256];
  red[threadIdx.x] = ss;
  __syncthreads();
  for (int off = 128; off; off >>= 1) {
    if ((int)threadIdx.x < off) red[threadIdx.x] += red[threadIdx.x + off];
    __syncthreads();
  }
  const float sc = g[o] / fmaxf(sqrtf(red[0]), 1e-8f);
  for (int i = threadIdx.x; i < R; i += 256) {
    const int c = i / kk, j = i % kk;
    w[(size_t)o * R + (size_t)j * Cin + c] = vr[i] * sc;
  }
}

// weight-norm for conv_transpose: v (C,O=C,4), norm per in-channel c.
// Output TRANSPOSED to (O, 4, C).
__global__ void k_wn_ups(const float* __restrict__ v, const float* __restrict__ g,
                         float* __restrict__ w, int C) {
  const int c = blockIdx.x;
  const int R = C * 4;
  const float* vr = v + (size_t)c * R;
  float ss = 0.0f;
  for (int i = threadIdx.x; i < R; i += 256) ss += vr[i] * vr[i];
  __shared__ float red[256];
  red[threadIdx.x] = ss;
  __syncthreads();
  for (int off = 128; off; off >>= 1) {
    if ((int)threadIdx.x < off) red[threadIdx.x] += red[threadIdx.x + off];
    __syncthreads();
  }
  const float sc = g[c] / fmaxf(sqrtf(red[0]), 1e-8f);
  for (int i = threadIdx.x; i < R; i += 256) {
    const int o = i >> 2, j = i & 3;
    w[((size_t)o * 4 + j) * C + c] = vr[i] * sc;
  }
}

// causal conv: X(B,T,Cin) -> Y(B,T,O); Wn in (O,kk,Cin) layout; left pad kk-1
__global__ void k_conv(const float* __restrict__ X, const float* __restrict__ Wn,
                       float* __restrict__ Y, int Bn, int T, int Cin, int O, int kk) {
  const size_t idx = (size_t)blockIdx.x * 256 + threadIdx.x;
  const size_t tot = (size_t)Bn * T * O;
  if (idx >= tot) return;
  const int o = (int)(idx % O);
  const int t = (int)((idx / O) % T);
  const int b = (int)(idx / ((size_t)O * T));
  float s = 0.0f;
  for (int j = 0; j < kk; ++j) {
    const int tp = t - (kk - 1) + j;
    if (tp < 0) continue;
    const float* xr = X + ((size_t)b * T + tp) * Cin;
    const float* wr = Wn + ((size_t)o * kk + j) * Cin;
    for (int c = 0; c < Cin; ++c) s += xr[c] * wr[c];
  }
  Y[((size_t)b * T + t) * O + o] = s;
}

// conv_transpose1d stride=2, k=4; Wn in (O,4,C) layout; out (B,2T,C), tail trimmed
__global__ void k_ups(const float* __restrict__ X, const float* __restrict__ Wn,
                      float* __restrict__ Y, int Bn, int Tin, int C) {
  const size_t idx = (size_t)blockIdx.x * 256 + threadIdx.x;
  const int Tout = 2 * Tin;
  const size_t tot = (size_t)Bn * Tout * C;
  if (idx >= tot) return;
  const int o = (int)(idx % C);
  const int t = (int)((idx / C) % Tout);
  const int b = (int)(idx / ((size_t)C * Tout));
  float s = 0.0f;
  for (int j = 0; j < 4; ++j) {
    const int ti = t - j;
    if (ti < 0 || (ti & 1)) continue;
    const int i = ti >> 1;
    if (i >= Tin) continue;
    const float* xr = X + ((size_t)b * Tin + i) * C;
    const float* wr = Wn + ((size_t)o * 4 + j) * C;
    for (int c = 0; c < C; ++c) s += xr[c] * wr[c];
  }
  Y[((size_t)b * Tout + t) * C + o] = s;
}

// sliding-window causal attention + ALiBi; one wave per (b,t,h); window<=16
__global__ void k_attn(const float* __restrict__ q, const float* __restrict__ k,
                       const float* __restrict__ v, float* __restrict__ o,
                       int Bn, int T, int win) {
  const int lane = threadIdx.x & 31;
  const int gw = (int)((blockIdx.x * (size_t)blockDim.x + threadIdx.x) >> 5);
  const int h = gw % NH;
  const int t = (gw / NH) % T;
  const int b = gw / (NH * T);
  if (b >= Bn) return;

  const float* qp = q + ((size_t)b * T + t) * DIMC + h * HDIM;
  float qr[4];
#pragma unroll
  for (int i = 0; i < 4; ++i) qr[i] = qp[lane + 32 * i];

  int j0 = t - win; if (j0 < 0) j0 = 0;
  const int nj = t - j0 + 1;                 // <= 17
  const float slope = exp2f(-(float)(h + 1));
  float s[17];
  for (int jj = 0; jj < nj; ++jj) {
    const int j = j0 + jj;
    const float* kp = k + ((size_t)b * T + j) * DIMC + h * HDIM;
    float d = 0.0f;
#pragma unroll
    for (int i = 0; i < 4; ++i) d += qr[i] * kp[lane + 32 * i];
    for (int off = 16; off; off >>= 1) d += __shfl_xor(d, off, 32);
    s[jj] = d * 0.08838834764831845f + slope * (float)(j - t);
  }
  float mx = -1e30f;
  for (int jj = 0; jj < nj; ++jj) mx = fmaxf(mx, s[jj]);
  float den = 0.0f;
  for (int jj = 0; jj < nj; ++jj) { s[jj] = __expf(s[jj] - mx); den += s[jj]; }
  const float inv = 1.0f / den;
  float acc[4] = {0.f, 0.f, 0.f, 0.f};
  for (int jj = 0; jj < nj; ++jj) {
    const int j = j0 + jj;
    const float* vp = v + ((size_t)b * T + j) * DIMC + h * HDIM;
    const float a = s[jj] * inv;
#pragma unroll
    for (int i = 0; i < 4; ++i) acc[i] += a * vp[lane + 32 * i];
  }
  float* op = o + ((size_t)b * T + t) * DIMC + h * HDIM;
#pragma unroll
  for (int i = 0; i < 4; ++i) op[lane + 32 * i] = acc[i];
}

__global__ void k_cvt_bf(const float* __restrict__ x, __bf16* __restrict__ y, size_t n) {
  const size_t i = (size_t)blockIdx.x * 256 + threadIdx.x;
  if (i < n) y[i] = f2bf(x[i]);
}

__global__ void k_silu_mul(const __bf16* __restrict__ a, const __bf16* __restrict__ b3,
                           __bf16* __restrict__ y, size_t n) {
  const size_t i = (size_t)blockIdx.x * 256 + threadIdx.x;
  if (i >= n) return;
  const float a0 = bf2f(a[i]);
  const float g = a0 / (1.0f + __expf(-a0));
  y[i] = f2bf(g * bf2f(b3[i]));
}

__global__ void k_resadd(float* __restrict__ x, const float* __restrict__ y,
                         const float* __restrict__ scale, size_t n) {
  const size_t i = (size_t)blockIdx.x * 256 + threadIdx.x;
  if (i >= n) return;
  const int c = (int)(i % DIMC);
  x[i] += scale[c] * y[i];
}

__global__ void k_embed(const int* __restrict__ codes, const float* __restrict__ tab,
                        const float* __restrict__ usage, float* __restrict__ y, int T) {
  const size_t idx = (size_t)blockIdx.x * 256 + threadIdx.x;
  const int C = SEMD + NACC;
  const size_t tot = (size_t)BATCH * T * C;
  if (idx >= tot) return;
  const int j = (int)(idx % C);
  const int t = (int)((idx / C) % T);
  const int b = (int)(idx / ((size_t)C * T));
  const int* cr = codes + ((size_t)b * T + t) * 37;
  float val;
  if (j < SEMD) {
    int si = cr[0] - 2;
    si = si < 0 ? 0 : (si > SEMN - 1 ? SEMN - 1 : si);
    val = tab[(size_t)si * SEMD + j] / fmaxf(usage[si], 1e-8f);
  } else {
    val = (float)(cr[j - SEMD + 1] - 2) * 0.1f - 1.0f;   // 2/(FSQ-1) = 0.1
  }
  y[idx] = val;
}

// ---------------------------------------------------------------------------
extern "C" void kernel_launch(void* const* d_in, const int* in_sizes, int n_in,
                              void* d_out, int out_size, void* d_ws, size_t ws_size,
                              hipStream_t stream) {
  (void)in_sizes; (void)n_in; (void)out_size; (void)ws_size;

  const int*   codes     = (const int*)d_in[0];
  const float* sem_table = (const float*)d_in[1];
  const float* sem_usage = (const float*)d_in[2];
  const float* in_v      = (const float*)d_in[3];
  const float* in_g      = (const float*)d_in[4];
  const float* ups_v[3]; const float* ups_g[3];
  for (int i = 0; i < 3; ++i) {
    ups_v[i] = (const float*)d_in[5 + 2 * i];
    ups_g[i] = (const float*)d_in[6 + 2 * i];
  }
  const float* out_v = (const float*)d_in[11];
  const float* out_g = (const float*)d_in[12];
  // stages start at 13: per layer 13 tensors: wq wk wv wo qn kn an fn ascale fscale w1 w2 w3

  // ---- workspace carve ----
  char* p = (char*)d_ws;
  auto carve = [&](size_t bytes) -> void* {
    void* r = (void*)p;
    p += (bytes + 255) & ~(size_t)255;
    return r;
  };
  const size_t ACT = (size_t)BATCH * 2048 * DIMC;
  const size_t HN  = (size_t)BATCH * 2048 * HIDC;
  float*  x0   = (float*)carve(ACT * 4);
  float*  x1   = (float*)carve(ACT * 4);
  float*  qb   = (float*)carve(ACT * 4);
  float*  kb   = (float*)carve(ACT * 4);
  float*  vb   = (float*)carve(ACT * 4);
  float*  att  = (float*)carve(ACT * 4);
  float*  cw   = (float*)carve((size_t)DIMC * DIMC * 4 * 4);
  float*  emb  = (float*)carve((size_t)BATCH * TT0 * (SEMD + NACC) * 4);
  __bf16* nrmbf = (__bf16*)carve(ACT * 2);
  __bf16* h1    = (__bf16*)carve(HN * 2);
  __bf16* h3    = (__bf16*)carve(HN * 2);
  __bf16* hb    = (__bf16*)carve(HN * 2);
  __bf16* wbf   = (__bf16*)carve((size_t)HIDC * DIMC * 2);

  auto blocks = [](size_t n) { return (unsigned)((n + 255) / 256); };
  auto tobf = [&](const float* src, __bf16* dst, size_t n) {
    k_cvt_bf<<<blocks(n), 256, 0, stream>>>(src, dst, n);
  };
  auto gemm = [&](const __bf16* A, const __bf16* Wt, void* C, int M, int N, int K,
                  bool obf) {
    dim3 g((unsigned)(N / 256), (unsigned)(M / 32));
    if (obf) k_gemm_wmma<1><<<g, 256, 0, stream>>>(A, Wt, C, M, N, K);
    else     k_gemm_wmma<0><<<g, 256, 0, stream>>>(A, Wt, C, M, N, K);
  };

  // ---- embed + input causal conv (k=3, Cin=292 -> 1024) ----
  k_embed<<<blocks((size_t)BATCH * TT0 * (SEMD + NACC)), 256, 0, stream>>>(
      codes, sem_table, sem_usage, emb, TT0);
  k_wn_oc<<<DIMC, 256, 0, stream>>>(in_v, in_g, cw, SEMD + NACC, 3);
  k_conv<<<blocks((size_t)BATCH * TT0 * DIMC), 256, 0, stream>>>(
      emb, cw, x0, BATCH, TT0, SEMD + NACC, DIMC, 3);

  float* x = x0;
  float* xalt = x1;
  const int Ts[4] = {256, 512, 1024, 2048};
  const int WS[4] = {2, 4, 8, 16};

  for (int s = 0; s < 4; ++s) {
    const int T = Ts[s];
    const int M = BATCH * T;
    const int win = WS[s];
    for (int l = 0; l < 2; ++l) {
      const int bi = 13 + (s * 2 + l) * 13;
      const float* wq = (const float*)d_in[bi + 0];
      const float* wk = (const float*)d_in[bi + 1];
      const float* wv = (const float*)d_in[bi + 2];
      const float* wo = (const float*)d_in[bi + 3];
      const float* qn = (const float*)d_in[bi + 4];
      const float* kn = (const float*)d_in[bi + 5];
      const float* an = (const float*)d_in[bi + 6];
      const float* fn = (const float*)d_in[bi + 7];
      const float* ascale = (const float*)d_in[bi + 8];
      const float* fscale = (const float*)d_in[bi + 9];
      const float* w1 = (const float*)d_in[bi + 10];
      const float* w2 = (const float*)d_in[bi + 11];
      const float* w3 = (const float*)d_in[bi + 12];

      // ---- attention block ----
      k_rms<1><<<M, 256, 0, stream>>>(x, an, nrmbf, DIMC);
      tobf(wq, wbf, (size_t)DIMC * DIMC);
      gemm(nrmbf, wbf, qb, M, DIMC, DIMC, false);
      k_rms<0><<<M, 256, 0, stream>>>(qb, qn, qb, DIMC);
      tobf(wk, wbf, (size_t)DIMC * DIMC);
      gemm(nrmbf, wbf, kb, M, DIMC, DIMC, false);
      k_rms<0><<<M, 256, 0, stream>>>(kb, kn, kb, DIMC);
      tobf(wv, wbf, (size_t)DIMC * DIMC);
      gemm(nrmbf, wbf, vb, M, DIMC, DIMC, false);
      {
        const size_t waves = (size_t)M * NH;
        k_attn<<<blocks(waves * 32), 256, 0, stream>>>(qb, kb, vb, att, BATCH, T, win);
      }
      tobf(att, nrmbf, (size_t)M * DIMC);
      tobf(wo, wbf, (size_t)DIMC * DIMC);
      gemm(nrmbf, wbf, att, M, DIMC, DIMC, false);
      k_resadd<<<blocks((size_t)M * DIMC), 256, 0, stream>>>(x, att, ascale,
                                                             (size_t)M * DIMC);

      // ---- FFN block (SwiGLU) ----
      k_rms<1><<<M, 256, 0, stream>>>(x, fn, nrmbf, DIMC);
      tobf(w1, wbf, (size_t)HIDC * DIMC);
      gemm(nrmbf, wbf, h1, M, HIDC, DIMC, true);
      tobf(w3, wbf, (size_t)HIDC * DIMC);
      gemm(nrmbf, wbf, h3, M, HIDC, DIMC, true);
      k_silu_mul<<<blocks((size_t)M * HIDC), 256, 0, stream>>>(h1, h3, hb,
                                                               (size_t)M * HIDC);
      tobf(w2, wbf, (size_t)DIMC * HIDC);
      gemm(hb, wbf, att, M, DIMC, HIDC, false);
      k_resadd<<<blocks((size_t)M * DIMC), 256, 0, stream>>>(x, att, fscale,
                                                             (size_t)M * DIMC);
    }
    if (s < 3) {
      k_wn_ups<<<DIMC, 256, 0, stream>>>(ups_v[s], ups_g[s], cw, DIMC);
      k_ups<<<blocks((size_t)BATCH * 2 * T * DIMC), 256, 0, stream>>>(
          x, cw, xalt, BATCH, T, DIMC);
      float* tswap = x; x = xalt; xalt = tswap;
    }
  }

  // ---- output causal conv (k=7, 1024 -> 240), writes d_out in (B, T*240) order
  k_wn_oc<<<PATCHC, 256, 0, stream>>>(out_v, out_g, cw, DIMC, 7);
  k_conv<<<blocks((size_t)BATCH * 2048 * PATCHC), 256, 0, stream>>>(
      x, cw, (float*)d_out, BATCH, 2048, DIMC, PATCHC, 7);
}
